// My_multihead_attention_43662637531565
// MI455X (gfx1250) — matrix operand order repaired and checked by reading, not compiled, not run on hardware
//
#include <hip/hip_runtime.h>
#include <cstdint>
#include <cstddef>

typedef unsigned short bf16_t;  // raw bf16 storage
typedef __attribute__((ext_vector_type(16))) __bf16 v16bf;
typedef __attribute__((ext_vector_type(8)))  __bf16 v8bf;
typedef __attribute__((ext_vector_type(8)))  float  v8f;

#define D_MODEL  1024
#define NUM_HEAD 16
#define HEAD_DIM 64
#define SEQ      2048
#define BATCH    2
#define NROWS    (BATCH * SEQ)  // 4096

union Frag16 { v16bf v; v8bf h[2]; };

__device__ inline v8f zero8() {
  v8f z;
#pragma unroll
  for (int i = 0; i < 8; i++) z[i] = 0.0f;
  return z;
}

__device__ inline bf16_t f2bf(float f) {  // round-to-nearest-even f32 -> bf16
  unsigned u = __float_as_uint(f);
  u += 0x7FFFu + ((u >> 16) & 1u);
  return (bf16_t)(u >> 16);
}

__device__ inline v8bf ldg8(const bf16_t* p) { return *(const v8bf*)p; }

__device__ inline float rmax16(float v) {  // reduce across 16-lane half
#pragma unroll
  for (int m = 1; m < 16; m <<= 1) v = fmaxf(v, __shfl_xor(v, m, 32));
  return v;
}
__device__ inline float rsum16(float v) {
#pragma unroll
  for (int m = 1; m < 16; m <<= 1) v += __shfl_xor(v, m, 32);
  return v;
}

__device__ inline v8f wmma_bf16(v16bf a, v16bf b, v8f c) {
  return __builtin_amdgcn_wmma_f32_16x16x32_bf16(
      /*neg_a=*/false, a, /*neg_b=*/false, b,
      /*c_mod=*/(short)0, c, /*reuse_a=*/false, /*reuse_b=*/false);
}

// ---------------------------------------------------------------------------
// Stage 0: fp32 -> bf16 conversion
// ---------------------------------------------------------------------------
__global__ void cvt_f32_to_bf16(const float* __restrict__ src,
                                bf16_t* __restrict__ dst, int n) {
  int i = blockIdx.x * blockDim.x + threadIdx.x;
  if (i < n) dst[i] = f2bf(src[i]);
}

// ---------------------------------------------------------------------------
// Stage 1: fused QKV projection GEMM (x @ w^T) + RoPE on Q,K.
// Each wave computes a 16(M) x 64(N) tile; block = 8 waves = 8 M-tiles.
// Double-buffered register stages so next-k loads overlap current WMMAs.
// Outputs: Qb,Kb as [B,H,S,64] bf16 ; V transposed as Vt [B,H,64,S] bf16.
// ---------------------------------------------------------------------------
__global__ __launch_bounds__(256) void qkv_rope_kernel(
    const bf16_t* __restrict__ xb, const bf16_t* __restrict__ wqb,
    const bf16_t* __restrict__ wkb, const bf16_t* __restrict__ wvb,
    bf16_t* __restrict__ Qb, bf16_t* __restrict__ Kb,
    bf16_t* __restrict__ Vt) {
  const int lane = threadIdx.x & 31;
  const int wave = threadIdx.x >> 5;
  const int lm = lane & 15, h8 = lane >> 4;
  const int mtile = blockIdx.x * 8 + wave;
  const int rowbase = mtile * 16;
  const int widx = blockIdx.y >> 4;            // 0:Q 1:K 2:V
  const int nbase = (blockIdx.y & 15) * 64;
  const bf16_t* w = (widx == 0) ? wqb : (widx == 1 ? wkb : wvb);

  const bf16_t* arow = xb + (size_t)(rowbase + lm) * D_MODEL;
  const bf16_t* wcol = w + (size_t)(nbase + lm) * D_MODEL + h8 * 8;

  v8f acc[4];
#pragma unroll
  for (int t = 0; t < 4; t++) acc[t] = zero8();

  Frag16 a[2], b[2][4];
  auto load_stage = [&](int st, int k) {
    a[st].h[0] = ldg8(arow + k + h8 * 8);
    a[st].h[1] = ldg8(arow + k + 16 + h8 * 8);
#pragma unroll
    for (int t = 0; t < 4; t++) {
      const bf16_t* brow = wcol + (size_t)(t * 16) * D_MODEL + k;
      b[st][t].h[0] = ldg8(brow);
      b[st][t].h[1] = ldg8(brow + 16);
    }
  };

  load_stage(0, 0);
#pragma unroll 2
  for (int k = 0; k < D_MODEL; k += 32) {
    const int cur = (k >> 5) & 1;
    if (k + 32 < D_MODEL) {
      __builtin_prefetch(arow + k + 160, 0, 0);  // stream-ahead hint
      load_stage(cur ^ 1, k + 32);
    }
#pragma unroll
    for (int t = 0; t < 4; t++)
      acc[t] = wmma_bf16(a[cur].v, b[cur][t].v, acc[t]);
  }

  const float kln = 9.210340371976184f / 64.0f;  // ln(10000)/HEAD_DIM
#pragma unroll
  for (int t = 0; t < 4; t++) {
    const int n = nbase + t * 16 + lm;
    const int head = n >> 6, d = n & 63;
    const int i2 = d & ~1;                       // 2*i of the RoPE pair
#pragma unroll
    for (int r = 0; r < 8; r++) {
      const int row = rowbase + r + 8 * h8;
      const int bb = row >> 11, s = row & (SEQ - 1);
      float v = acc[t][r];
      if (widx < 2) {                            // RoPE: partner in lane^1
        float p = __shfl_xor(v, 1, 32);
        float fr = (float)s * __expf(-(float)i2 * kln);
        float sn, cs;
        __sincosf(fr, &sn, &cs);
        v = (d & 1) ? (p * sn + v * cs) : (v * cs - p * sn);
        bf16_t* dst = (widx == 0) ? Qb : Kb;
        dst[(((size_t)(bb * NUM_HEAD + head)) * SEQ + s) * HEAD_DIM + d] =
            f2bf(v);
      } else {                                   // V stored transposed
        Vt[(((size_t)(bb * NUM_HEAD + head)) * HEAD_DIM + d) * SEQ + s] =
            f2bf(v);
      }
    }
  }
}

// ---------------------------------------------------------------------------
// Stage 2: causal flash attention. One wave per (b, head, 16-query tile).
// Keys streamed in chunks of 32 with online softmax; P re-laid via LDS.
// V fragments hoisted before the softmax VALU block to hide their latency.
// Output merged heads as bf16 [B*S, D_MODEL].
// ---------------------------------------------------------------------------
__global__ __launch_bounds__(32) void flash_attn_kernel(
    const bf16_t* __restrict__ Qb, const bf16_t* __restrict__ Kb,
    const bf16_t* __restrict__ Vt, bf16_t* __restrict__ attnb) {
  const int lane = threadIdx.x;
  const int lm = lane & 15, h8 = lane >> 4;
  const int qt = blockIdx.x, head = blockIdx.y, b = blockIdx.z;
  const int qbase = qt * 16;

  const bf16_t* Qh = Qb + ((size_t)(b * NUM_HEAD + head)) * SEQ * HEAD_DIM;
  const bf16_t* Kh = Kb + ((size_t)(b * NUM_HEAD + head)) * SEQ * HEAD_DIM;
  const bf16_t* Vh = Vt + ((size_t)(b * NUM_HEAD + head)) * HEAD_DIM * SEQ;

  __shared__ __align__(16) bf16_t Plds[16 * 32];

  Frag16 aq[2];                                  // Q A-frags, d in [0,32) [32,64)
  {
    const bf16_t* qrow = Qh + (size_t)(qbase + lm) * HEAD_DIM;
#pragma unroll
    for (int c = 0; c < 2; c++) {
      aq[c].h[0] = ldg8(qrow + c * 32 + h8 * 8);
      aq[c].h[1] = ldg8(qrow + c * 32 + 16 + h8 * 8);
    }
  }

  float mrow[8], lrow[8];
  v8f acc[4];
#pragma unroll
  for (int r = 0; r < 8; r++) { mrow[r] = -3.0e38f; lrow[r] = 0.0f; }
#pragma unroll
  for (int t = 0; t < 4; t++) acc[t] = zero8();

  const float scale = 0.125f;                    // 1/sqrt(64)
  const int kend = qbase + 16;                   // causal bound

  for (int k0 = 0; k0 < kend; k0 += 32) {
    v8f sc[2];                                   // scores: two 16x16 tiles
#pragma unroll
    for (int j = 0; j < 2; j++) {
      v8f c = zero8();
#pragma unroll
      for (int cd = 0; cd < 2; cd++) {           // K^T is B-operand: row=key
        const bf16_t* krow =
            Kh + (size_t)(k0 + j * 16 + lm) * HEAD_DIM + cd * 32 + h8 * 8;
        Frag16 bk;
        bk.h[0] = ldg8(krow);
        bk.h[1] = ldg8(krow + 16);
        c = wmma_bf16(aq[cd].v, bk.v, c);
      }
      sc[j] = c;
    }

    // V B-frags issued now: latency hidden behind the softmax VALU block
    Frag16 bv[4];
#pragma unroll
    for (int t = 0; t < 4; t++) {
      const bf16_t* vrow = Vh + (size_t)(t * 16 + lm) * SEQ + k0 + h8 * 8;
      bv[t].h[0] = ldg8(vrow);
      bv[t].h[1] = ldg8(vrow + 16);
    }

    // online softmax (row statistics live per half-wave in D layout)
#pragma unroll
    for (int r = 0; r < 8; r++) {
      const int q = qbase + r + 8 * h8;
      float s0 = sc[0][r] * scale; if (k0 + lm      > q) s0 = -1.0e30f;
      float s1 = sc[1][r] * scale; if (k0 + 16 + lm > q) s1 = -1.0e30f;
      float mx = rmax16(fmaxf(s0, s1));
      float mn = fmaxf(mrow[r], mx);
      float p0 = __expf(s0 - mn);
      float p1 = __expf(s1 - mn);
      float rs = rsum16(p0 + p1);
      float cr = __expf(mrow[r] - mn);
      lrow[r] = lrow[r] * cr + rs;
      mrow[r] = mn;
#pragma unroll
      for (int t = 0; t < 4; t++) acc[t][r] *= cr;
      const int prow = r + 8 * h8;
      Plds[prow * 32 + lm]      = f2bf(p0);      // reshape P via LDS
      Plds[prow * 32 + 16 + lm] = f2bf(p1);
    }
    __syncthreads();

    Frag16 ap;                                   // P as A-operand (16x32)
    ap.h[0] = *(const v8bf*)&Plds[lm * 32 + h8 * 8];
    ap.h[1] = *(const v8bf*)&Plds[lm * 32 + 16 + h8 * 8];
    __syncthreads();

#pragma unroll
    for (int t = 0; t < 4; t++)
      acc[t] = wmma_bf16(ap.v, bv[t].v, acc[t]);
  }

#pragma unroll
  for (int r = 0; r < 8; r++) {
    const float inv = 1.0f / lrow[r];
    const size_t row = (size_t)b * SEQ + qbase + r + 8 * h8;
#pragma unroll
    for (int t = 0; t < 4; t++) {
      attnb[row * D_MODEL + head * HEAD_DIM + t * 16 + lm] =
          f2bf(acc[t][r] * inv);
    }
  }
}

// ---------------------------------------------------------------------------
// Stage 3: output projection (attn @ wo^T) -> fp32 d_out, double-buffered
// ---------------------------------------------------------------------------
__global__ __launch_bounds__(256) void out_proj_kernel(
    const bf16_t* __restrict__ attnb, const bf16_t* __restrict__ wob,
    float* __restrict__ out) {
  const int lane = threadIdx.x & 31;
  const int wave = threadIdx.x >> 5;
  const int lm = lane & 15, h8 = lane >> 4;
  const int rowbase = (blockIdx.x * 8 + wave) * 16;
  const int nbase = blockIdx.y * 64;

  const bf16_t* arow = attnb + (size_t)(rowbase + lm) * D_MODEL;
  const bf16_t* wcol = wob + (size_t)(nbase + lm) * D_MODEL + h8 * 8;

  v8f acc[4];
#pragma unroll
  for (int t = 0; t < 4; t++) acc[t] = zero8();

  Frag16 a[2], b[2][4];
  auto load_stage = [&](int st, int k) {
    a[st].h[0] = ldg8(arow + k + h8 * 8);
    a[st].h[1] = ldg8(arow + k + 16 + h8 * 8);
#pragma unroll
    for (int t = 0; t < 4; t++) {
      const bf16_t* brow = wcol + (size_t)(t * 16) * D_MODEL + k;
      b[st][t].h[0] = ldg8(brow);
      b[st][t].h[1] = ldg8(brow + 16);
    }
  };

  load_stage(0, 0);
#pragma unroll 2
  for (int k = 0; k < D_MODEL; k += 32) {
    const int cur = (k >> 5) & 1;
    if (k + 32 < D_MODEL) {
      __builtin_prefetch(arow + k + 160, 0, 0);
      load_stage(cur ^ 1, k + 32);
    }
#pragma unroll
    for (int t = 0; t < 4; t++)
      acc[t] = wmma_bf16(a[cur].v, b[cur][t].v, acc[t]);
  }
#pragma unroll
  for (int t = 0; t < 4; t++) {
#pragma unroll
    for (int r = 0; r < 8; r++) {
      out[(size_t)(rowbase + r + 8 * h8) * D_MODEL + nbase + t * 16 + lm] =
          acc[t][r];
    }
  }
}

// ---------------------------------------------------------------------------
extern "C" void kernel_launch(void* const* d_in, const int* in_sizes, int n_in,
                              void* d_out, int out_size, void* d_ws,
                              size_t ws_size, hipStream_t stream) {
  (void)in_sizes; (void)n_in; (void)out_size; (void)ws_size;
  const float* x  = (const float*)d_in[0];
  const float* wq = (const float*)d_in[1];
  const float* wk = (const float*)d_in[2];
  const float* wv = (const float*)d_in[3];
  const float* wo = (const float*)d_in[4];
  float* out = (float*)d_out;

  char* p = (char*)d_ws;
  auto carve = [&](size_t elems) -> bf16_t* {
    bf16_t* r = (bf16_t*)p;
    size_t bytes = (elems * sizeof(bf16_t) + 255) & ~(size_t)255;
    p += bytes;
    return r;
  };
  bf16_t* xb    = carve((size_t)NROWS * D_MODEL);
  bf16_t* wqb   = carve((size_t)D_MODEL * D_MODEL);
  bf16_t* wkb   = carve((size_t)D_MODEL * D_MODEL);
  bf16_t* wvb   = carve((size_t)D_MODEL * D_MODEL);
  bf16_t* wob   = carve((size_t)D_MODEL * D_MODEL);
  bf16_t* Qb    = carve((size_t)BATCH * NUM_HEAD * SEQ * HEAD_DIM);
  bf16_t* Kb    = carve((size_t)BATCH * NUM_HEAD * SEQ * HEAD_DIM);
  bf16_t* Vt    = carve((size_t)BATCH * NUM_HEAD * HEAD_DIM * SEQ);
  bf16_t* attnb = carve((size_t)NROWS * D_MODEL);

  {
    int n = NROWS * D_MODEL;
    cvt_f32_to_bf16<<<n / 256, 256, 0, stream>>>(x, xb, n);
    n = D_MODEL * D_MODEL;
    cvt_f32_to_bf16<<<n / 256, 256, 0, stream>>>(wq, wqb, n);
    cvt_f32_to_bf16<<<n / 256, 256, 0, stream>>>(wk, wkb, n);
    cvt_f32_to_bf16<<<n / 256, 256, 0, stream>>>(wv, wvb, n);
    cvt_f32_to_bf16<<<n / 256, 256, 0, stream>>>(wo, wob, n);
  }

  // 256 M-tiles / 8 per block = 32 ; 3*1024 cols / 64 = 48
  qkv_rope_kernel<<<dim3(32, 48), 256, 0, stream>>>(xb, wqb, wkb, wvb, Qb, Kb,
                                                    Vt);
  flash_attn_kernel<<<dim3(SEQ / 16, NUM_HEAD, BATCH), 32, 0, stream>>>(
      Qb, Kb, Vt, attnb);
  out_proj_kernel<<<dim3(32, 16), 256, 0, stream>>>(attnb, wob, out);
}
